// Layer_Attention_MultiHead_GATv2_48155173323387
// MI455X (gfx1250) — compile-verified, matrix operand order checked by reading, not run on hardware
//
#include <hip/hip_runtime.h>
#include <hip/hip_bf16.h>

typedef __attribute__((ext_vector_type(16))) _Float16 v16h;
typedef __attribute__((ext_vector_type(8)))  _Float16 v8h;
typedef __attribute__((ext_vector_type(4)))  _Float16 v4h;
typedef __attribute__((ext_vector_type(8)))  float    v8f;

#define GAT_N   4096
#define GAT_IN  128
#define GAT_R   64
#define GAT_H   8
#define GAT_C   512   // HEADS*REPR
#define LDS_STRIDE 40 // 32 halves + 8 pad (80B rows -> spread 16B lane loads)

// ---------------------------------------------------------------------------
// Kernel 1: hidden[j, h*64+l] = node[j,:] @ W[h,:,l]   and
//           f2[h,j] = leaky_relu(hidden[j,h,:]) . a_dst[h]
// grid = (N/4, HEADS), block = 256 (4 nodes x 64 lanes)
// ---------------------------------------------------------------------------
__global__ __launch_bounds__(256) void gat_hidden_f2(
    const float* __restrict__ node, const float* __restrict__ W,
    const float* __restrict__ attn_vec, float* __restrict__ hidden,
    float* __restrict__ f2) {
  __shared__ float nodeS[4][GAT_IN];
  __shared__ float red[256];
  const int tid = threadIdx.x;
  const int h = blockIdx.y;
  const int j0 = blockIdx.x * 4;
  for (int q = 0; q < 2; ++q) {
    int idx = tid + q * 256;                       // 0..511
    nodeS[idx >> 7][idx & 127] =
        node[(size_t)(j0 + (idx >> 7)) * GAT_IN + (idx & 127)];
  }
  __syncthreads();
  const int l = tid & 63;
  const int jl = tid >> 6;
  const float* wcol = W + (size_t)h * GAT_IN * GAT_R + l;
  float acc = 0.f;
#pragma unroll 8
  for (int k = 0; k < GAT_IN; ++k) acc += nodeS[jl][k] * wcol[(size_t)k * GAT_R];
  const int j = j0 + jl;
  hidden[(size_t)j * GAT_C + h * GAT_R + l] = acc;
  float act = acc > 0.f ? acc : 0.2f * acc;
  red[tid] = act * attn_vec[h * (2 * GAT_R) + GAT_R + l];   // a_dst
  __syncthreads();
  for (int s = 32; s >= 1; s >>= 1) {
    if ((tid & 63) < s) red[tid] += red[tid + s];
    __syncthreads();
  }
  if ((tid & 63) == 0) f2[h * GAT_N + j] = red[tid];
}

// ---------------------------------------------------------------------------
// Kernel 2: per-head max of f2 (softmax stabilization). grid = HEADS.
// ---------------------------------------------------------------------------
__global__ __launch_bounds__(256) void gat_rowmax(const float* __restrict__ f2,
                                                  float* __restrict__ m) {
  __shared__ float red[256];
  const int h = blockIdx.x, tid = threadIdx.x;
  float v = -3.0e38f;
  for (int k = tid; k < GAT_N; k += 256) v = fmaxf(v, f2[h * GAT_N + k]);
  red[tid] = v;
  __syncthreads();
  for (int s = 128; s >= 1; s >>= 1) {
    if (tid < s) red[tid] = fmaxf(red[tid], red[tid + s]);
    __syncthreads();
  }
  if (tid == 0) m[h] = red[0];
}

// ---------------------------------------------------------------------------
// Kernel 3: e[h,k] = exp(f2 - m[h]).  grid = H*N/256.
// ---------------------------------------------------------------------------
__global__ void gat_exp(const float* __restrict__ f2, const float* __restrict__ m,
                        float* __restrict__ e) {
  const int idx = blockIdx.x * 256 + threadIdx.x;   // 0..H*N-1
  const int h = idx >> 12;
  e[idx] = __expf(f2[idx] - m[h]);
}

// ---------------------------------------------------------------------------
// Kernel 4: Bt[col, k] = f16( e[h,k] * hidden[k, col] ), col = h*64+l.
// Stored col-major in k (Bt[col*N + k]) so GEMM B-tile loads are contiguous.
// ---------------------------------------------------------------------------
__global__ void gat_bt(const float* __restrict__ hidden, const float* __restrict__ e,
                       _Float16* __restrict__ Bt) {
  const size_t idx = (size_t)blockIdx.x * 256 + threadIdx.x;  // 0..C*N-1
  const int k = (int)(idx & (GAT_N - 1));
  const int col = (int)(idx >> 12);
  const int h = col >> 6;
  Bt[idx] = (_Float16)(e[h * GAT_N + k] * hidden[(size_t)k * GAT_C + col]);
}

// ---------------------------------------------------------------------------
// Kernel 5: denom[h,j] = sum_k adj[j,k] * e[h,k].  grid = N, block = 256.
// ---------------------------------------------------------------------------
__global__ __launch_bounds__(256) void gat_denom(const float* __restrict__ adj,
                                                 const float* __restrict__ e,
                                                 float* __restrict__ denom) {
  __shared__ float red[256];
  const int j = blockIdx.x, tid = threadIdx.x;
  float acc[GAT_H] = {};
  for (int k = tid; k < GAT_N; k += 256) {
    const float a = adj[(size_t)j * GAT_N + k];
    if (a != 0.f) {
#pragma unroll
      for (int h = 0; h < GAT_H; ++h) acc[h] += a * e[h * GAT_N + k];
    }
  }
  for (int h = 0; h < GAT_H; ++h) {
    red[tid] = acc[h];
    __syncthreads();
    for (int s = 128; s >= 1; s >>= 1) {
      if (tid < s) red[tid] += red[tid + s];
      __syncthreads();
    }
    if (tid == 0) denom[h * GAT_N + j] = red[0];
    __syncthreads();
  }
}

// ---------------------------------------------------------------------------
// Kernel 6: C(4096x512) = adj(f32->f16, {0,1} exact) @ Bt, fused /denom.
// 128x128 C tile per block, 4 waves (128 threads); each wave owns a 32x128
// strip via TWO A fragments, so every B fragment feeds two WMMAs:
// per K-step: 20 ds_load_b128 -> 16 v_wmma (vs 18 -> 8 before), halving
// LDS bytes per WMMA — the f16 XDL pipe outruns LDS at 1 fresh B/WMMA.
// ---------------------------------------------------------------------------
__global__ __launch_bounds__(128) void gat_gemm(
    const float* __restrict__ adj, const _Float16* __restrict__ Bt,
    const float* __restrict__ denom, float* __restrict__ out) {
  __shared__ _Float16 lds_a[128 * LDS_STRIDE];
  __shared__ _Float16 lds_b[128 * LDS_STRIDE];
  const int tid = threadIdx.x;
  const int wave = tid >> 5;             // 0..3, owns rows [wave*32, wave*32+32)
  const int lane = tid & 31;
  const int row0 = blockIdx.y * 128;
  const int col0 = blockIdx.x * 128;
  const int r = lane & 15;               // fragment row/col within 16x16 tile
  const int kbase = (lane >> 4) << 3;    // 0 or 8 (ISA 16-bit A/B layout)

  v8f acc0[8], acc1[8];
#pragma unroll
  for (int n = 0; n < 8; ++n) { acc0[n] = (v8f)(0.0f); acc1[n] = (v8f)(0.0f); }

  for (int kb = 0; kb < GAT_N; kb += 32) {
    // Stage A tile: 128x32 f32 adjacency -> f16 in LDS (values are exact 0/1)
#pragma unroll
    for (int q = 0; q < 8; ++q) {
      const int e4 = tid + q * 128;         // float4 index 0..1023
      const int ar = e4 >> 3;               // 8 float4 per 32-wide row
      const int ak = (e4 & 7) << 2;
      const float4 v =
          *(const float4*)(adj + (size_t)(row0 + ar) * GAT_N + kb + ak);
      v4h hv = { (_Float16)v.x, (_Float16)v.y, (_Float16)v.z, (_Float16)v.w };
      *(v4h*)(&lds_a[ar * LDS_STRIDE + ak]) = hv;
    }
    // Stage B tile: 128 cols x 32 k halves (contiguous in k)
#pragma unroll
    for (int q = 0; q < 4; ++q) {
      const int e8 = tid + q * 128;         // half8 index 0..511
      const int bc = e8 >> 2;
      const int bk = (e8 & 3) << 3;
      v8h hv = *(const v8h*)(Bt + (size_t)(col0 + bc) * GAT_N + kb + bk);
      *(v8h*)(&lds_b[bc * LDS_STRIDE + bk]) = hv;
    }
    __syncthreads();

    // Two A fragments: rows wave*32 + r and wave*32 + 16 + r
    const _Float16* pa0 = &lds_a[(wave * 32 + r) * LDS_STRIDE + kbase];
    const _Float16* pa1 = pa0 + 16 * LDS_STRIDE;
    const v8h a0lo = *(const v8h*)(pa0);
    const v8h a0hi = *(const v8h*)(pa0 + 16);
    const v8h a1lo = *(const v8h*)(pa1);
    const v8h a1hi = *(const v8h*)(pa1 + 16);
    const v16h a0 = __builtin_shufflevector(a0lo, a0hi, 0, 1, 2, 3, 4, 5, 6, 7,
                                            8, 9, 10, 11, 12, 13, 14, 15);
    const v16h a1 = __builtin_shufflevector(a1lo, a1hi, 0, 1, 2, 3, 4, 5, 6, 7,
                                            8, 9, 10, 11, 12, 13, 14, 15);
#pragma unroll
    for (int n = 0; n < 8; ++n) {
      const _Float16* pb = &lds_b[(n * 16 + r) * LDS_STRIDE + kbase];
      const v8h blo = *(const v8h*)(pb);
      const v8h bhi = *(const v8h*)(pb + 16);
      const v16h b = __builtin_shufflevector(blo, bhi, 0, 1, 2, 3, 4, 5, 6, 7,
                                             8, 9, 10, 11, 12, 13, 14, 15);
      acc0[n] = __builtin_amdgcn_wmma_f32_16x16x32_f16(
          false, a0, false, b, (short)0, acc0[n], false, false);
      acc1[n] = __builtin_amdgcn_wmma_f32_16x16x32_f16(
          false, a1, false, b, (short)0, acc1[n], false, false);
    }
    __syncthreads();
  }

  // Epilogue: C layout — VGPR i of lane L holds (M = i + 8*(L/16), Ncol = L%16)
  const int mbase = (lane >> 4) << 3;
  const int ncol = lane & 15;
#pragma unroll
  for (int n = 0; n < 8; ++n) {
    const int col = col0 + n * 16 + ncol;
    const int h = col >> 6;
#pragma unroll
    for (int i = 0; i < 8; ++i) {
      const int row_a = row0 + wave * 32 + mbase + i;
      const int row_b = row_a + 16;
      out[(size_t)row_a * GAT_C + col] = acc0[n][i] / denom[h * GAT_N + row_a];
      out[(size_t)row_b * GAT_C + col] = acc1[n][i] / denom[h * GAT_N + row_b];
    }
  }
}

// ---------------------------------------------------------------------------
extern "C" void kernel_launch(void* const* d_in, const int* in_sizes, int n_in,
                              void* d_out, int out_size, void* d_ws,
                              size_t ws_size, hipStream_t stream) {
  const float* node = (const float*)d_in[0];   // (4096,128)
  const float* adj  = (const float*)d_in[1];   // (4096,4096)
  const float* W    = (const float*)d_in[2];   // (8,128,64)
  const float* av   = (const float*)d_in[3];   // (8,128)
  float* out = (float*)d_out;                  // (4096,512)

  char* ws = (char*)d_ws;
  float*    hidden = (float*)(ws);                                // 8 MB
  float*    f2     = (float*)(ws + (size_t)8 * 1024 * 1024);      // 128 KB
  float*    m      = (float*)(ws + (size_t)8 * 1024 * 1024 + 131072);        // 256 B
  float*    e      = (float*)(ws + (size_t)8 * 1024 * 1024 + 131072 + 256);  // 128 KB
  float*    denom  = (float*)(ws + (size_t)8 * 1024 * 1024 + 2 * 131072 + 256);
  _Float16* Bt     = (_Float16*)(ws + (size_t)8 * 1024 * 1024 + 3 * 131072 + 256); // 4 MB

  gat_hidden_f2<<<dim3(GAT_N / 4, GAT_H), 256, 0, stream>>>(node, W, av, hidden, f2);
  gat_rowmax<<<GAT_H, 256, 0, stream>>>(f2, m);
  gat_exp<<<(GAT_H * GAT_N) / 256, 256, 0, stream>>>(f2, m, e);
  gat_bt<<<(GAT_C * GAT_N) / 256, 256, 0, stream>>>(hidden, e, Bt);
  gat_denom<<<GAT_N, 256, 0, stream>>>(adj, e, denom);
  gat_gemm<<<dim3(GAT_C / 128, GAT_N / 128), 128, 0, stream>>>(adj, Bt, denom, out);
}
